// Attention_cross_audio_visual_37726992728225
// MI455X (gfx1250) — compile-verified, matrix operand order checked by reading, not run on hardware
//
#include <hip/hip_runtime.h>

typedef __bf16 bf16;
typedef __attribute__((ext_vector_type(16))) __bf16 v16bf;
typedef __attribute__((ext_vector_type(8)))  __bf16 v8bf;
typedef __attribute__((ext_vector_type(8)))  float  v8f;

// ---------------------------------------------------------------- helpers

__device__ inline v8f vzero() {
  v8f z;
#pragma unroll
  for (int i = 0; i < 8; ++i) z[i] = 0.0f;
  return z;
}

__device__ inline v8f wmma_bf16(v16bf a, v16bf b, v8f c) {
  // D = A(16x32) * B(32x16) + C, f32 accumulate
  return __builtin_amdgcn_wmma_f32_16x16x32_bf16(
      /*neg_a=*/false, a, /*neg_b=*/false, b,
      /*c_mod=*/(short)0, c, /*reuse_a=*/false, /*reuse_b=*/false);
}

// Load a 16x32 bf16 A-style fragment from row-major storage (row stride `ld`
// elements). Per ISA layout: lane L holds row M=L&15; half-wave selects K
// chunks {half*8..+7} and {16+half*8..+7} -> two contiguous 16B loads.
__device__ inline v16bf ldfrag(const bf16* base, int ld) {
  int lane = threadIdx.x & 31;
  int m    = lane & 15;
  int half = lane >> 4;
  const bf16* p = base + (long)m * ld + half * 8;
  v8bf lo = *(const v8bf*)(p);
  v8bf hi = *(const v8bf*)(p + 16);
  v16bf f;
#pragma unroll
  for (int i = 0; i < 8; ++i) { f[i] = lo[i]; f[i + 8] = hi[i]; }
  return f;
}

// ---------------------------------------------------------------- convert

__global__ __launch_bounds__(256) void cvt_k(const float* __restrict__ s,
                                             bf16* __restrict__ d, int n) {
  int i = blockIdx.x * 256 + threadIdx.x;
  if (i < n) d[i] = (bf16)s[i];
}

// ---------------------------------------------------------------- GEMM
// C[M,N] = A[M,K] * B[K,N] (+bias).  128x128 block tile, 256 threads
// (8 waves, each owning a 32x64 region = 8 WMMA tiles), K-step 32.
// bf16 A tiles are staged via gfx1250 async global->LDS (ASYNCcnt);
// f32 A tiles are converted to bf16 during manual staging; B is staged
// transposed (sBt[n][k]) so B fragments are contiguous 16B chunks.
template <bool A_F32, bool C_F32, bool BIAS>
__global__ __launch_bounds__(256) void gemm_k(
    const void* __restrict__ A, long ldA,
    const bf16* __restrict__ B, int ldB,
    const float* __restrict__ bias,
    void* __restrict__ C, long ldC, int K) {
  __shared__ alignas(16) bf16 sA [128][40];
  __shared__ alignas(16) bf16 sBt[128][40];

  const long m0 = (long)blockIdx.x * 128;
  const long n0 = (long)blockIdx.y * 128;
  const int tid  = threadIdx.x;
  const int w    = tid >> 5;
  const int lane = tid & 31;
  const int wm   = w & 3;    // wave row    -> m tiles wm*2 .. wm*2+1
  const int wn   = w >> 2;   // wave column -> n tiles wn*4 .. wn*4+3

  v8f acc[2][4];
#pragma unroll
  for (int i = 0; i < 2; ++i)
#pragma unroll
    for (int j = 0; j < 4; ++j) acc[i][j] = vzero();

  // staging maps: A: 128 rows x 32 k, 16 elts/thread.  B: 32 k x 128 n.
  const int ar = tid >> 1, ac = (tid & 1) * 16;
  const int bk = tid >> 3, bn = (tid & 7) * 16;

  for (int k0 = 0; k0 < K; k0 += 32) {
    if (A_F32) {
      const float* Ap = (const float*)A + (m0 + ar) * ldA + k0 + ac;
      if (k0 + 32 < K) __builtin_prefetch(Ap + 32, 0, 1);
#pragma unroll
      for (int i = 0; i < 16; ++i) sA[ar][ac + i] = (bf16)Ap[i];
    } else {
      const bf16* Ap = (const bf16*)A + (m0 + ar) * ldA + k0 + ac;
      if (k0 + 32 < K) __builtin_prefetch(Ap + 32, 0, 1);
      // gfx1250 async copy: 2 x 16B per thread, LDS dest from VGPR,
      // IOFFSET applies to both global and LDS sides (ISA 08 §4.4).
      auto* lp = (__attribute__((address_space(3))) bf16*)&sA[ar][ac];
      unsigned int       ldsa = (unsigned int)(unsigned long long)lp;
      unsigned long long ga   = (unsigned long long)Ap;
      asm volatile("global_load_async_to_lds_b128 %0, %1, off"
                   :: "v"(ldsa), "v"(ga) : "memory");
      asm volatile("global_load_async_to_lds_b128 %0, %1, off offset:16"
                   :: "v"(ldsa), "v"(ga) : "memory");
    }
    { // stage B tile 32x128 transposed -> sBt[n][k]
      const bf16* Bp = B + (long)(k0 + bk) * ldB + n0 + bn;
      if (k0 + 32 < K) __builtin_prefetch(Bp + 32 * ldB, 0, 1);
      v8bf x0 = *(const v8bf*)(Bp);
      v8bf x1 = *(const v8bf*)(Bp + 8);
#pragma unroll
      for (int i = 0; i < 8; ++i) {
        sBt[bn + i][bk]     = x0[i];
        sBt[bn + 8 + i][bk] = x1[i];
      }
    }
    if (!A_F32)
      asm volatile("s_wait_asynccnt 0x0" ::: "memory");
    __syncthreads();

    v16bf fa0 = ldfrag(&sA[(wm * 2 + 0) * 16][0], 40);
    v16bf fa1 = ldfrag(&sA[(wm * 2 + 1) * 16][0], 40);
#pragma unroll
    for (int j = 0; j < 4; ++j) {
      v16bf fb = ldfrag(&sBt[(wn * 4 + j) * 16][0], 40);
      acc[0][j] = wmma_bf16(fa0, fb, acc[0][j]);
      acc[1][j] = wmma_bf16(fa1, fb, acc[1][j]);
    }
    __syncthreads();
  }

  const int lm = (lane >> 4) * 8, ln = lane & 15;
#pragma unroll
  for (int i = 0; i < 2; ++i)
#pragma unroll
    for (int j = 0; j < 4; ++j)
#pragma unroll
      for (int r = 0; r < 8; ++r) {
        long gm = m0 + (wm * 2 + i) * 16 + r + lm;
        long gn = n0 + (wn * 4 + j) * 16 + ln;
        float v = acc[i][j][r];
        if (BIAS) v += bias[gn];
        if (C_F32) ((float*)C)[gm * ldC + gn] = v;
        else       ((bf16*)C)[gm * ldC + gn] = (bf16)v;
      }
}

// ---------------------------------------------------------------- attention
// One wave handles one (b,h): S = softmax(scale * Q K^T) held as 2x2 WMMA
// accumulators.  QKV layout: qkv[row=b*32+n][qkv_i*768 + h*96 + d], bf16.

__device__ inline void attn_softmax(const bf16* __restrict__ qkv, long b, int h,
                                    float scale, v8f S[2][2]) {
  const bf16* Qb = qkv + b * 32 * 2304 + (long)h * 96;
  const bf16* Kb = Qb + 768;
  v8f z = vzero();
  S[0][0] = z; S[0][1] = z; S[1][0] = z; S[1][1] = z;
#pragma unroll
  for (int kc = 0; kc < 3; ++kc) {       // head_dim 96 = 3 * 32
    v16bf q0 = ldfrag(Qb + kc * 32, 2304);
    v16bf q1 = ldfrag(Qb + 16 * 2304 + kc * 32, 2304);
    v16bf k0 = ldfrag(Kb + kc * 32, 2304);   // K^T frag = contiguous K rows
    v16bf k1 = ldfrag(Kb + 16 * 2304 + kc * 32, 2304);
    S[0][0] = wmma_bf16(q0, k0, S[0][0]);
    S[0][1] = wmma_bf16(q0, k1, S[0][1]);
    S[1][0] = wmma_bf16(q1, k0, S[1][0]);
    S[1][1] = wmma_bf16(q1, k1, S[1][1]);
  }
  // row softmax: row stats live in a 16-lane half-group per (ti, r)
#pragma unroll
  for (int ti = 0; ti < 2; ++ti) {
#pragma unroll
    for (int r = 0; r < 8; ++r) {
      float a = S[ti][0][r] * scale;
      float c = S[ti][1][r] * scale;
      float mx = fmaxf(a, c);
#pragma unroll
      for (int off = 1; off < 16; off <<= 1)
        mx = fmaxf(mx, __shfl_xor(mx, off, 32));
      float ea = __expf(a - mx), ec = __expf(c - mx);
      float s = ea + ec;
#pragma unroll
      for (int off = 1; off < 16; off <<= 1)
        s += __shfl_xor(s, off, 32);
      float inv = 1.0f / s;
      S[ti][0][r] = ea * inv;
      S[ti][1][r] = ec * inv;
    }
  }
}

// Pass 1: accumulate mean attention maps per (batch, head) via f32 atomics.
__global__ __launch_bounds__(32) void attn_mean_k(const bf16* __restrict__ qkv,
                                                  float* __restrict__ meanbuf,
                                                  int patches, float scale) {
  long bh = blockIdx.x;
  int  h  = (int)(bh & 7);
  long b  = bh >> 3;
  v8f S[2][2];
  attn_softmax(qkv, b, h, scale, S);
  long batch = b / patches;
  float wgt = 1.0f / (float)patches;
  float* mb = meanbuf + (batch * 8 + h) * 1024;
  int lane = threadIdx.x;
  int lm = (lane >> 4) * 8, ln = lane & 15;
#pragma unroll
  for (int ti = 0; ti < 2; ++ti)
#pragma unroll
    for (int tj = 0; tj < 2; ++tj)
#pragma unroll
      for (int r = 0; r < 8; ++r)
        __hip_atomic_fetch_add(&mb[(ti * 16 + r + lm) * 32 + tj * 16 + ln],
                               S[ti][tj][r] * wgt,
                               __ATOMIC_RELAXED, __HIP_MEMORY_SCOPE_AGENT);
}

// Pass 2: P = 0.9*S + 0.1*mean_other;  O = P @ V, written over the Q section.
__global__ __launch_bounds__(32) void attn_out_k(bf16* __restrict__ qkv,
                                                 const float* __restrict__ meanOther,
                                                 int patches, float scale) {
  __shared__ alignas(16) float sP [32][33];
  __shared__ alignas(16) bf16  sVt[96][40];
  long bh = blockIdx.x;
  int  h  = (int)(bh & 7);
  long b  = bh >> 3;
  v8f S[2][2];
  attn_softmax(qkv, b, h, scale, S);
  long batch = b / patches;
  const float* mo = meanOther + (batch * 8 + h) * 1024;
  int lane = threadIdx.x;
  int lm = (lane >> 4) * 8, ln = lane & 15;
#pragma unroll
  for (int ti = 0; ti < 2; ++ti)
#pragma unroll
    for (int tj = 0; tj < 2; ++tj)
#pragma unroll
      for (int r = 0; r < 8; ++r) {
        int M = ti * 16 + r + lm, N = tj * 16 + ln;
        sP[M][N] = 0.9f * S[ti][tj][r] + 0.1f * mo[M * 32 + N];
      }
  // stage V (32x96) transposed: sVt[d][k]
  const bf16* Vb = qkv + b * 32 * 2304 + 1536 + (long)h * 96;
#pragma unroll
  for (int c = 0; c < 12; ++c) {
    v8bf v = *(const v8bf*)(Vb + (long)lane * 2304 + c * 8);
#pragma unroll
    for (int i = 0; i < 8; ++i) sVt[c * 8 + i][lane] = v[i];
  }
  __syncthreads();

  bf16* Ob = qkv + b * 32 * 2304 + (long)h * 96;   // overwrite consumed Q
  int m = lane & 15, half = lane >> 4;
#pragma unroll
  for (int ti = 0; ti < 2; ++ti) {
    v16bf fa;
    const float* pr = &sP[ti * 16 + m][0];
#pragma unroll
    for (int i = 0; i < 8; ++i) {
      fa[i]     = (bf16)pr[half * 8 + i];
      fa[i + 8] = (bf16)pr[16 + half * 8 + i];
    }
#pragma unroll
    for (int j = 0; j < 6; ++j) {                  // 96 = 6 * 16 output cols
      v16bf fb = ldfrag(&sVt[j * 16][0], 40);
      v8f acc = vzero();
      acc = wmma_bf16(fa, fb, acc);
#pragma unroll
      for (int r = 0; r < 8; ++r)
        Ob[(long)(ti * 16 + r + lm) * 2304 + j * 16 + ln] = (bf16)acc[r];
    }
  }
}

// ---------------------------------------------------------------- launch

extern "C" void kernel_launch(void* const* d_in, const int* in_sizes, int n_in,
                              void* d_out, int out_size, void* d_ws, size_t ws_size,
                              hipStream_t stream) {
  const float* vis       = (const float*)d_in[0];
  const float* aud       = (const float*)d_in[1];
  const float* Wqkv_vis  = (const float*)d_in[2];
  const float* Wproj_vis = (const float*)d_in[3];
  const float* bproj_vis = (const float*)d_in[4];
  const float* Wqkv_aud  = (const float*)d_in[5];
  const float* Wproj_aud = (const float*)d_in[6];
  const float* bproj_aud = (const float*)d_in[7];

  const long MV = 50176;   // 8 * 196 * 32 rows
  const long MA = 16384;   // 8 *  64 * 32 rows

  char* base = (char*)d_ws;
  size_t off = 0;
  auto alloc = [&](size_t bytes) -> char* {
    char* p = base + off;
    off = (off + bytes + 255) & ~(size_t)255;
    return p;
  };
  bf16*  qkv_vis = (bf16*)alloc(MV * 2304 * 2);
  bf16*  qkv_aud = (bf16*)alloc(MA * 2304 * 2);
  bf16*  wqv     = (bf16*)alloc((size_t)768 * 2304 * 2);
  bf16*  wqa     = (bf16*)alloc((size_t)768 * 2304 * 2);
  bf16*  wpv     = (bf16*)alloc((size_t)768 * 768 * 2);
  bf16*  wpa     = (bf16*)alloc((size_t)768 * 768 * 2);
  float* means   = (float*)alloc((size_t)2 * 65536 * 4);
  float* mean_vis = means;            // pooled vis maps (consumed by aud side)
  float* mean_aud = means + 65536;    // pooled aud maps (consumed by vis side)

  // 1. weights fp32 -> bf16
  cvt_k<<<(768 * 2304 + 255) / 256, 256, 0, stream>>>(Wqkv_vis, wqv, 768 * 2304);
  cvt_k<<<(768 * 2304 + 255) / 256, 256, 0, stream>>>(Wqkv_aud, wqa, 768 * 2304);
  cvt_k<<<(768 * 768  + 255) / 256, 256, 0, stream>>>(Wproj_vis, wpv, 768 * 768);
  cvt_k<<<(768 * 768  + 255) / 256, 256, 0, stream>>>(Wproj_aud, wpa, 768 * 768);
  hipMemsetAsync(means, 0, (size_t)2 * 65536 * 4, stream);

  // 2. QKV GEMMs (f32 A converted in-LDS, bf16 out)
  gemm_k<true, false, false><<<dim3(MV / 128, 2304 / 128), 256, 0, stream>>>(
      vis, 768, wqv, 2304, nullptr, qkv_vis, 2304, 768);
  gemm_k<true, false, false><<<dim3(MA / 128, 2304 / 128), 256, 0, stream>>>(
      aud, 768, wqa, 2304, nullptr, qkv_aud, 2304, 768);

  const float scale = 0.10206207261596577f;   // (768/8)^-0.5

  // 3. pooled attention maps
  attn_mean_k<<<MV / 32 * 8, 32, 0, stream>>>(qkv_vis, mean_vis, 196, scale);
  attn_mean_k<<<MA / 32 * 8, 32, 0, stream>>>(qkv_aud, mean_aud, 64, scale);

  // 4. mixed attention * V -> O (overwrites Q section)
  attn_out_k<<<MV / 32 * 8, 32, 0, stream>>>(qkv_vis, mean_aud, 196, scale);
  attn_out_k<<<MA / 32 * 8, 32, 0, stream>>>(qkv_aud, mean_vis, 64, scale);

  // 5. output projections (+bias, async-LDS A staging) -> f32 d_out
  float* out_vis = (float*)d_out;
  float* out_aud = out_vis + MV * 768;
  gemm_k<false, true, true><<<dim3(MV / 128, 768 / 128), 256, 0, stream>>>(
      qkv_vis, 2304, wpv, 768, bproj_vis, out_vis, 768, 768);
  gemm_k<false, true, true><<<dim3(MA / 128, 768 / 128), 256, 0, stream>>>(
      qkv_aud, 2304, wpa, 768, bproj_aud, out_aud, 768, 768);
}
